// MIGCProcessor_30408368456380
// MI455X (gfx1250) — compile-verified, matrix operand order checked by reading, not run on hardware
//
#include <hip/hip_runtime.h>
#include <hip/hip_bf16.h>
#include <math.h>

typedef __attribute__((ext_vector_type(16))) __bf16 v16bf;
typedef __attribute__((ext_vector_type(8)))  float  v8f;

#define B_SZ   2
#define N_TOK  2304
#define C_DIM  1280
#define HEADS  8
#define DH     160
#define SCALE  0.07905694150420949f   /* 1/sqrt(160) */
#define LOG2E  1.4426950408889634f

// ---------- helpers ----------
// Native bf16 convert (v_cvt on gfx1250) instead of manual 4-op rounding.
__device__ __forceinline__ unsigned short f2bf(float f) {
  union { __bf16 h; unsigned short u; } c;
  c.h = (__bf16)f;
  return c.u;
}

// Load a 16x32 bf16 A/B fragment from a row-major LDS tile.
// A layout: lanes 0-15 -> M=0-15, VGPR v packs K={2v,2v+1} (v<4) or K={2v+8,2v+9} (v>=4); lanes 16-31: K+8.
// From row-major [row][k] storage this is two contiguous 16B chunks per lane.
__device__ __forceinline__ v16bf ld_frag(const unsigned short* tile, int row,
                                         int pitch, int col, int lane) {
  const int r  = lane & 15;
  const int hi = lane >> 4;
  const unsigned short* p = tile + (size_t)(row + r) * pitch + col + hi * 8;
  union { int4 i[2]; v16bf v; } u;
  u.i[0] = *(const int4*)(p);        // K 0..7  (+8 for hi lanes)
  u.i[1] = *(const int4*)(p + 16);   // K 16..23 (+8 for hi lanes)
  return u.v;
}

// ======================= Kernel 1: projection GEMM =======================
// Y[b*H+h][n][d] (bf16) = (X @ W)[row=b*N+n][col=h*DH+d], fp32 in, bf16 out.
// Block: 256 thr / 8 waves; tile 128(M) x 128(N); K-step 64 (16 WMMA / barrier).
__global__ __launch_bounds__(256)
void proj_gemm(const float* __restrict__ X, const float* __restrict__ W,
               unsigned short* __restrict__ Y) {
  __shared__ __align__(16) unsigned short sA[128 * 72];  // X tile, pitch 72
  __shared__ __align__(16) unsigned short sB[128 * 72];  // W^T tile [n][k]
  const int tid = threadIdx.x, lane = tid & 31, wave = tid >> 5;
  const int wm = wave & 1, wn = wave >> 1;               // 2x4 wave grid
  const int m0g = blockIdx.x * 128, n0g = blockIdx.y * 128;

  v8f acc[4][2];
#pragma unroll
  for (int i = 0; i < 4; ++i)
#pragma unroll
    for (int j = 0; j < 2; ++j) acc[i][j] = {};

  for (int kt = 0; kt < C_DIM / 64; ++kt) {
    const int kb = kt * 64;
    // stage A: 128x64 f32 -> bf16 (coalesced float4 loads, packed b64 stores)
#pragma unroll
    for (int i = 0; i < 8; ++i) {
      int m = tid + i * 256;                 // 2048 float4 total
      int row = m >> 4, c = m & 15;
      float4 f = *(const float4*)(X + (size_t)(m0g + row) * C_DIM + kb + c * 4);
      uint2 pk;
      pk.x = f2bf(f.x) | ((unsigned)f2bf(f.y) << 16);
      pk.y = f2bf(f.z) | ((unsigned)f2bf(f.w) << 16);
      *(uint2*)(sA + (size_t)row * 72 + c * 4) = pk;
    }
    // stage B transposed: sB[n][k] = W[kb+k][n0g+n]
    {
      int krow = tid >> 2;                   // 64 k-rows, 4 threads per row
#pragma unroll
      for (int p = 0; p < 8; ++p) {
        int n = p * 16 + (tid & 3) * 4;
        float4 f = *(const float4*)(W + (size_t)(kb + krow) * C_DIM + n0g + n);
        sB[(size_t)(n + 0) * 72 + krow] = f2bf(f.x);
        sB[(size_t)(n + 1) * 72 + krow] = f2bf(f.y);
        sB[(size_t)(n + 2) * 72 + krow] = f2bf(f.z);
        sB[(size_t)(n + 3) * 72 + krow] = f2bf(f.w);
      }
    }
    __syncthreads();
#pragma unroll
    for (int kc = 0; kc < 2; ++kc) {
      v16bf a[4], b[2];
#pragma unroll
      for (int i = 0; i < 4; ++i)
        a[i] = ld_frag(sA, wm * 64 + i * 16, 72, kc * 32, lane);
#pragma unroll
      for (int j = 0; j < 2; ++j)
        b[j] = ld_frag(sB, wn * 32 + j * 16, 72, kc * 32, lane);
#pragma unroll
      for (int i = 0; i < 4; ++i)
#pragma unroll
        for (int j = 0; j < 2; ++j)
          acc[i][j] = __builtin_amdgcn_wmma_f32_16x16x32_bf16(
              false, a[i], false, b[j], (short)0, acc[i][j], false, false);
    }
    __syncthreads();
  }
  // epilogue: D layout -> head-major bf16 [bh][n][d]; h/d math hoisted out of v
  const int r = lane & 15, hi = lane >> 4;
  const size_t HSZ = (size_t)HEADS * N_TOK * DH;         // batch stride
#pragma unroll
  for (int i = 0; i < 4; ++i)
#pragma unroll
    for (int j = 0; j < 2; ++j) {
      const int col_g = n0g + wn * 32 + j * 16 + r;      // [0,1280)
      const int h  = col_g / DH;
      const int d  = col_g - h * DH;
      const size_t hoff = (size_t)h * N_TOK * DH + d;
      const int row_b = m0g + wm * 64 + i * 16 + hi * 8;
#pragma unroll
      for (int v = 0; v < 8; ++v) {
        int row_g = row_b + v;                           // [0,4608)
        int bb = row_g >= N_TOK;
        int nn = row_g - bb * N_TOK;
        Y[(size_t)bb * HSZ + (size_t)nn * DH + hoff] = f2bf(acc[i][j][v]);
      }
    }
}

// ======================= Kernel 2: flash attention =======================
// Block: 128 thr / 4 waves; 64 queries per block, key blocks of 32.
// Per wave: 16 queries. S^T = K*Q^T so P converts to A-frag by register packing.
__global__ __launch_bounds__(128)
void attn_fused(const unsigned short* __restrict__ Qb,
                const unsigned short* __restrict__ Kb,
                const unsigned short* __restrict__ Vb,
                const float* __restrict__ masks,
                float* __restrict__ out) {
  constexpr int QP = 168, KP = 168, VP = 40, MP = 36;
  __shared__ __align__(16) unsigned short sQ[64 * QP];   // 21504 B
  __shared__ __align__(16) unsigned short sK[32 * KP];   // 10752 B
  __shared__ __align__(16) unsigned short sV[DH * VP];   // Vt[d][key], 12800 B
  __shared__ __align__(16) float          sM[64 * MP];   //  9216 B

  const int tid = threadIdx.x, lane = tid & 31, wave = tid >> 5;
  const int bh = blockIdx.y, b = bh >> 3, h = bh & 7;
  const int q0 = blockIdx.x * 64;
  const size_t headoff = (size_t)bh * N_TOK * DH;
  const unsigned short* Qg = Qb + headoff + (size_t)q0 * DH;
  const unsigned short* Kg = Kb + headoff;
  const unsigned short* Vg = Vb + headoff;
  const float* Mg = masks + (size_t)b * N_TOK * N_TOK + (size_t)q0 * N_TOK;

  // stage Q tile (64 x 160 bf16), dword-granular coalesced
#pragma unroll
  for (int i = 0; i < 40; ++i) {
    int dw = tid + i * 128;
    int row = dw / 80, c = dw - row * 80;
    ((unsigned*)(sQ + (size_t)row * QP))[c] =
        ((const unsigned*)(Qg + (size_t)row * DH))[c];
  }
  __syncthreads();

  const int wq = wave * 16;
  const int r = lane & 15, hi = lane >> 4;
  v16bf bQ[5];                                   // loop-invariant Q B-frags
#pragma unroll
  for (int kc = 0; kc < 5; ++kc) bQ[kc] = ld_frag(sQ, wq, QP, kc * 32, lane);

  v8f O[10];
#pragma unroll
  for (int t = 0; t < 10; ++t) O[t] = {};
  float m_run = -3.0e38f, l_run = 0.0f;

  for (int kb = 0; kb < N_TOK; kb += 32) {
    __syncthreads();   // previous iteration's compute done before reload
    // stage K (row-major) and V (transposed) blocks
#pragma unroll
    for (int i = 0; i < 20; ++i) {
      int dw = tid + i * 128;
      int row = dw / 80, c = dw - row * 80;
      unsigned kv = ((const unsigned*)(Kg + (size_t)(kb + row) * DH))[c];
      ((unsigned*)(sK + (size_t)row * KP))[c] = kv;
      unsigned vv = ((const unsigned*)(Vg + (size_t)(kb + row) * DH))[c];
      sV[(size_t)(2 * c) * VP + row]     = (unsigned short)(vv & 0xFFFFu);
      sV[(size_t)(2 * c + 1) * VP + row] = (unsigned short)(vv >> 16);
    }
    // stage mask tile 64x32 f32
#pragma unroll
    for (int i = 0; i < 4; ++i) {
      int m = tid + i * 128;
      int q = m >> 3, c = m & 7;
      *(float4*)(sM + (size_t)q * MP + c * 4) =
          *(const float4*)(Mg + (size_t)q * N_TOK + kb + c * 4);
    }
    if (kb + 32 < N_TOK)
      __builtin_prefetch(Kg + (size_t)(kb + 32) * DH + (size_t)tid * 40, 0, 1);
    __syncthreads();

    // S^T = K(32x160) x Q^T : keys in VGPR rows, queries in lanes
    v8f st0 = {}, st1 = {};
#pragma unroll
    for (int kc = 0; kc < 5; ++kc) {
      v16bf aK0 = ld_frag(sK,  0, KP, kc * 32, lane);
      v16bf aK1 = ld_frag(sK, 16, KP, kc * 32, lane);
      st0 = __builtin_amdgcn_wmma_f32_16x16x32_bf16(false, aK0, false, bQ[kc],
                                                    (short)0, st0, false, false);
      st1 = __builtin_amdgcn_wmma_f32_16x16x32_bf16(false, aK1, false, bQ[kc],
                                                    (short)0, st1, false, false);
    }

    // mask (query = wq+r, key = VGPR index + hi*8 [+16 for st1])
    const float* mrow = sM + (size_t)(wq + r) * MP + hi * 8;
    float4 ma = *(const float4*)(mrow);
    float4 mb = *(const float4*)(mrow + 4);
    float4 mc = *(const float4*)(mrow + 16);
    float4 md = *(const float4*)(mrow + 20);
    float mk0[8] = {ma.x, ma.y, ma.z, ma.w, mb.x, mb.y, mb.z, mb.w};
    float mk1[8] = {mc.x, mc.y, mc.z, mc.w, md.x, md.y, md.z, md.w};

    float s0[8], s1[8], mx = -3.0e38f;
#pragma unroll
    for (int v = 0; v < 8; ++v) {
      s0[v] = fmaf(st0[v], SCALE, mk0[v]);
      s1[v] = fmaf(st1[v], SCALE, mk1[v]);
      mx = fmaxf(mx, fmaxf(s0[v], s1[v]));
    }
    mx = fmaxf(mx, __shfl_xor(mx, 16, 32));           // combine lane pair (q,q+16)
    float mnew  = fmaxf(m_run, mx);
    float alpha = __builtin_amdgcn_exp2f((m_run - mnew) * LOG2E);
    float p0[8], p1[8], lsum = 0.0f;
#pragma unroll
    for (int v = 0; v < 8; ++v) {
      p0[v] = __builtin_amdgcn_exp2f((s0[v] - mnew) * LOG2E);
      p1[v] = __builtin_amdgcn_exp2f((s1[v] - mnew) * LOG2E);
      lsum += p0[v] + p1[v];
    }
    lsum += __shfl_xor(lsum, 16, 32);
    l_run = l_run * alpha + lsum;
    m_run = mnew;

    // rescale O: per-query alpha broadcast (O rows are queries v+hi*8)
    float av[8];
#pragma unroll
    for (int v = 0; v < 8; ++v) av[v] = __shfl(alpha, v + hi * 8, 32);
#pragma unroll
    for (int t = 0; t < 10; ++t)
#pragma unroll
      for (int v = 0; v < 8; ++v) O[t][v] *= av[v];

    // pack P: S^T D-layout maps directly onto A-frag layout (keys packed 2/dword)
    union { v16bf v; unsigned short e[16]; } pf;
#pragma unroll
    for (int v = 0; v < 8; ++v) {
      pf.e[v]     = f2bf(p0[v]);   // keys hi*8 + 0..7
      pf.e[8 + v] = f2bf(p1[v]);   // keys 16 + hi*8 + 0..7
    }
    // O += P(16x32) x V(32x160), B-frags from Vt[d][key]
#pragma unroll
    for (int t = 0; t < 10; ++t) {
      v16bf bV = ld_frag(sV, t * 16, VP, 0, lane);
      O[t] = __builtin_amdgcn_wmma_f32_16x16x32_bf16(false, pf.v, false, bV,
                                                     (short)0, O[t], false, false);
    }
  }

  // normalize + write out[b][n][h*DH+d] fp32
  float rl = 1.0f / l_run;
  float rv[8];
#pragma unroll
  for (int v = 0; v < 8; ++v) rv[v] = __shfl(rl, v + hi * 8, 32);
#pragma unroll
  for (int t = 0; t < 10; ++t)
#pragma unroll
    for (int v = 0; v < 8; ++v) {
      int qg = q0 + wq + v + hi * 8;
      int d  = t * 16 + r;
      out[((size_t)b * N_TOK + qg) * C_DIM + h * DH + d] = O[t][v] * rv[v];
    }
}

// ======================= launcher =======================
extern "C" void kernel_launch(void* const* d_in, const int* in_sizes, int n_in,
                              void* d_out, int out_size, void* d_ws, size_t ws_size,
                              hipStream_t stream) {
  (void)in_sizes; (void)n_in; (void)out_size; (void)ws_size;
  const float* X  = (const float*)d_in[0];
  const float* Wq = (const float*)d_in[1];
  const float* Wk = (const float*)d_in[2];
  const float* Wv = (const float*)d_in[3];
  const float* Mk = (const float*)d_in[4];
  float* out = (float*)d_out;

  const size_t S = (size_t)B_SZ * N_TOK * C_DIM;   // elements per projection
  unsigned short* Qb = (unsigned short*)d_ws;
  unsigned short* Kb = Qb + S;
  unsigned short* Vb = Kb + S;

  dim3 g1((B_SZ * N_TOK) / 128, C_DIM / 128), b1(256);
  proj_gemm<<<g1, b1, 0, stream>>>(X, Wq, Qb);
  proj_gemm<<<g1, b1, 0, stream>>>(X, Wk, Kb);
  proj_gemm<<<g1, b1, 0, stream>>>(X, Wv, Vb);

  dim3 g2(N_TOK / 64, B_SZ * HEADS), b2(128);
  attn_fused<<<g2, b2, 0, stream>>>(Qb, Kb, Vb, Mk, out);
}
